// LSTM_13503377178715
// MI455X (gfx1250) — compile-verified, matrix operand order checked by reading, not run on hardware
//
#include <hip/hip_runtime.h>

// ---------------------------------------------------------------------------
// 2-layer LSTM for MI455X (gfx1250, wave32, WMMA).
//
// Per step/layer: gates(256x4096) = [x|h](256x2048) @ Wcat(2048x4096) + bias,
// computed with v_wmma_f32_16x16x32_bf16 (bf16 in, f32 accumulate).
// Weights are packed to bf16 once per launch (amortized over 128 steps; the
// 32MB of bf16 weights live in the 192MB L2 across the whole sequence).
// ---------------------------------------------------------------------------

typedef __attribute__((ext_vector_type(16))) __bf16 v16bf;
typedef __attribute__((ext_vector_type(8)))  float  v8f;

#define S_LEN  128
#define BATCH  256
#define IN_DIM 1024
#define HID    1024
#define NG     4096   // 4*H gate width
#define KCAT   2048   // IN + H (or H + H)

// GEMM tiling
#define BM  64
#define BN  128
#define BK  32
#define LDK 40        // padded bf16 row stride (32 + 8) -> 80B rows, 16B aligned

union Pack8 { __bf16 h[8]; uint4 u; };

static __device__ __forceinline__ void cvt8(const float4 f0, const float4 f1, Pack8& p) {
    p.h[0] = (__bf16)f0.x; p.h[1] = (__bf16)f0.y; p.h[2] = (__bf16)f0.z; p.h[3] = (__bf16)f0.w;
    p.h[4] = (__bf16)f1.x; p.h[5] = (__bf16)f1.y; p.h[6] = (__bf16)f1.z; p.h[7] = (__bf16)f1.w;
}

// ---------------------------------------------------------------------------
// gates(256x4096) = [A0|A1](256x2048,f32) x W(N=4096 x K=2048, bf16, row-major)
//                   + bias, f32 out.
// Block: 256 threads = 8 waves, tile 64x128. Wave tile 32x32 (2x2 WMMA tiles).
// ---------------------------------------------------------------------------
__global__ __launch_bounds__(256) void lstm_gates_gemm(
    const float*  __restrict__ A0,    // (256,1024) K-half 0
    const float*  __restrict__ A1,    // (256,1024) K-half 1
    const __bf16* __restrict__ W,     // (4096,2048) row-major (N x K)
    const float*  __restrict__ bias,  // (4096)
    float*        __restrict__ gates) // (256,4096)
{
    __shared__ __bf16 As[2][BM][LDK];
    __shared__ __bf16 Bs[2][BN][LDK];

    const int tid  = threadIdx.x;
    const int bm   = blockIdx.x * BM;
    const int bn   = blockIdx.y * BN;
    const int wid  = tid >> 5;
    const int lane = tid & 31;
    const int l    = lane & 15;
    const int hx   = lane >> 4;        // lane-half selector (ISA 16-bit layouts)
    const int wm   = (wid >> 2) << 5;  // 0 / 32
    const int wn   = (wid & 3) << 5;   // 0,32,64,96

    // staging coordinates
    const int arow = tid >> 2;         // 0..63
    const int acol = (tid & 3) << 3;   // 0,8,16,24  (8 f32 -> 8 bf16)
    const int brow = tid >> 1;         // 0..127
    const int bcol = (tid & 1) << 4;   // 0,16       (16 bf16 = 32B)

    // accumulators, bias-initialized per C layout: VGPR r -> row r+8*hx, col l
    v8f acc[2][2];
#pragma unroll
    for (int tn = 0; tn < 2; ++tn) {
        const float bv = bias[bn + wn + tn * 16 + l];
        v8f t;
#pragma unroll
        for (int r = 0; r < 8; ++r) t[r] = bv;
        acc[0][tn] = t; acc[1][tn] = t;
    }

    // ---- prologue: stage chunk 0 (kc=0 is always in A0) ----
    {
        const float* src = A0 + (size_t)(bm + arow) * IN_DIM + acol;
        float4 f0 = *(const float4*)src;
        float4 f1 = *(const float4*)(src + 4);
        Pack8 p; cvt8(f0, f1, p);
        *(uint4*)&As[0][arow][acol] = p.u;

        const __bf16* wsrc = W + (size_t)(bn + brow) * KCAT + bcol;
        *(uint4*)&Bs[0][brow][bcol]     = *(const uint4*)(wsrc);
        *(uint4*)&Bs[0][brow][bcol + 8] = *(const uint4*)(wsrc + 8);
    }
    __syncthreads();

    const int nch = KCAT / BK; // 64
#pragma unroll 2
    for (int ch = 0; ch < nch; ++ch) {
        const int buf = ch & 1;

        // ---- stage next chunk into the other buffer ----
        if (ch + 1 < nch) {
            const int kc = (ch + 1) * BK;
            const float* abase = (kc < IN_DIM) ? A0 : A1;
            const int    koff  = (kc < IN_DIM) ? kc : (kc - IN_DIM);
            const float* src = abase + (size_t)(bm + arow) * IN_DIM + koff + acol;
            float4 f0 = *(const float4*)src;
            float4 f1 = *(const float4*)(src + 4);
            Pack8 p; cvt8(f0, f1, p);
            *(uint4*)&As[buf ^ 1][arow][acol] = p.u;

            const __bf16* wsrc = W + (size_t)(bn + brow) * KCAT + kc + bcol;
            uint4 w0 = *(const uint4*)(wsrc);
            uint4 w1 = *(const uint4*)(wsrc + 8);
            *(uint4*)&Bs[buf ^ 1][brow][bcol]     = w0;
            *(uint4*)&Bs[buf ^ 1][brow][bcol + 8] = w1;

            if (ch + 2 < nch)  // global_prefetch_b8 for the chunk after next
                __builtin_prefetch(W + (size_t)(bn + brow) * KCAT + (size_t)(ch + 2) * BK + bcol, 0, 1);
        }

        // ---- compute on buf: fragments per ISA 16-bit A/B lane layout ----
        // lane<16: K {0..7} then {16..23}; lane>=16: K {8..15} then {24..31}
        v16bf afr[2], bfr[2];
#pragma unroll
        for (int tm = 0; tm < 2; ++tm) {
            const __bf16* rp = &As[buf][wm + tm * 16 + l][0];
            union { v16bf v; uint4 u[2]; } fa;
            fa.u[0] = *(const uint4*)(rp + hx * 8);
            fa.u[1] = *(const uint4*)(rp + 16 + hx * 8);
            afr[tm] = fa.v;
        }
#pragma unroll
        for (int tn = 0; tn < 2; ++tn) {
            const __bf16* rp = &Bs[buf][wn + tn * 16 + l][0];
            union { v16bf v; uint4 u[2]; } fb;
            fb.u[0] = *(const uint4*)(rp + hx * 8);
            fb.u[1] = *(const uint4*)(rp + 16 + hx * 8);
            bfr[tn] = fb.v;
        }
#pragma unroll
        for (int tm = 0; tm < 2; ++tm)
#pragma unroll
            for (int tn = 0; tn < 2; ++tn)
                acc[tm][tn] = __builtin_amdgcn_wmma_f32_16x16x32_bf16(
                    false, afr[tm], false, bfr[tn], (short)0, acc[tm][tn], false, false);

        __syncthreads();
    }

    // ---- store: VGPR r -> row (base + r + 8*hx), col = l ----
#pragma unroll
    for (int tm = 0; tm < 2; ++tm) {
#pragma unroll
        for (int tn = 0; tn < 2; ++tn) {
            const int row0 = bm + wm + tm * 16 + hx * 8;
            const int col  = bn + wn + tn * 16 + l;
            float* gp = gates + (size_t)row0 * NG + col;
#pragma unroll
            for (int r = 0; r < 8; ++r)
                gp[(size_t)r * NG] = acc[tm][tn][r];
        }
    }
}

// ---------------------------------------------------------------------------
// Elementwise LSTM cell + reset masking (f32 throughout).
// ---------------------------------------------------------------------------
static __device__ __forceinline__ float sigf(float x) {
    return 1.0f / (1.0f + __expf(-x));
}
static __device__ __forceinline__ float tanh_fast(float x) {
    float ax = fabsf(x);
    float e  = __expf(-2.0f * ax);
    float t  = (1.0f - e) / (1.0f + e);
    return copysignf(t, x);
}

__global__ __launch_bounds__(256) void lstm_cell(
    const float* __restrict__ gates,  // (256,4096)
    const float* __restrict__ c_in,   // (256,1024)
    const int*   __restrict__ resets, // (256) for this step
    float* __restrict__ h_raw,        // (256,1024) pre-mask h (next-layer input / output)
    float* __restrict__ h_st,         // (256,1024) masked carry h
    float* __restrict__ c_st)         // (256,1024) masked carry c
{
    const int t = blockIdx.x * blockDim.x + threadIdx.x;  // 65536 threads, 4 elems each
    const int b = t >> 8;
    const int n = (t & 255) << 2;
    const float keep = 1.0f - (float)resets[b];

    const float* g = gates + (size_t)b * NG;
    float4 iv = *(const float4*)(g + n);
    float4 fv = *(const float4*)(g + HID + n);
    float4 gv = *(const float4*)(g + 2 * HID + n);
    float4 ov = *(const float4*)(g + 3 * HID + n);
    float4 cv = *(const float4*)(c_in + (size_t)b * HID + n);

    float ii[4] = {iv.x, iv.y, iv.z, iv.w};
    float ff[4] = {fv.x, fv.y, fv.z, fv.w};
    float gg[4] = {gv.x, gv.y, gv.z, gv.w};
    float oo[4] = {ov.x, ov.y, ov.z, ov.w};
    float cp[4] = {cv.x, cv.y, cv.z, cv.w};
    float hr[4], hc[4], cc[4];
#pragma unroll
    for (int k = 0; k < 4; ++k) {
        float i  = sigf(ii[k]);
        float f  = sigf(ff[k]);
        float gA = tanh_fast(gg[k]);
        float o  = sigf(oo[k]);
        float cn = f * cp[k] + i * gA;
        float hn = o * tanh_fast(cn);
        hr[k] = hn;
        hc[k] = hn * keep;
        cc[k] = cn * keep;
    }
    const size_t off = (size_t)b * HID + n;
    *(float4*)(h_raw + off) = make_float4(hr[0], hr[1], hr[2], hr[3]);
    *(float4*)(h_st  + off) = make_float4(hc[0], hc[1], hc[2], hc[3]);
    *(float4*)(c_st  + off) = make_float4(cc[0], cc[1], cc[2], cc[3]);
}

// ---------------------------------------------------------------------------
// One-time prep kernels.
// ---------------------------------------------------------------------------
__global__ __launch_bounds__(256) void pack_weights(
    const float* __restrict__ wih,  // (4096,1024)
    const float* __restrict__ whh,  // (4096,1024)
    __bf16* __restrict__ out)       // (4096,2048) row-major N x K
{
    const int t = blockIdx.x * 256 + threadIdx.x;  // 1,048,576 threads x 8 elems
    const int n = t >> 8;
    const int k = (t & 255) << 3;
    const float* src = (k < IN_DIM) ? (wih + (size_t)n * IN_DIM + k)
                                    : (whh + (size_t)n * HID + (k - IN_DIM));
    float4 f0 = *(const float4*)src;
    float4 f1 = *(const float4*)(src + 4);
    Pack8 p; cvt8(f0, f1, p);
    *(uint4*)(out + (size_t)n * KCAT + k) = p.u;
}

__global__ __launch_bounds__(256) void combine_bias(
    const float* __restrict__ bi, const float* __restrict__ bh, float* __restrict__ out)
{
    const int t = blockIdx.x * 256 + threadIdx.x;  // 4096
    out[t] = bi[t] + bh[t];
}

__global__ __launch_bounds__(256) void init_state(
    const float* __restrict__ mem,  // (2,256,2048)  [h | c]
    float* __restrict__ h0, float* __restrict__ c0,
    float* __restrict__ h1, float* __restrict__ c1)
{
    const int t = blockIdx.x * 256 + threadIdx.x;  // 1,048,576
    const int l = t >> 19;
    const int b = (t >> 11) & 255;
    const int j = t & 2047;
    const float v = mem[t];
    float* dst = (j < HID) ? (l ? h1 : h0) : (l ? c1 : c0);
    dst[(size_t)b * HID + (j & 1023)] = v;
}

__global__ __launch_bounds__(256) void finalize_state(
    const float* __restrict__ h0, const float* __restrict__ c0,
    const float* __restrict__ h1, const float* __restrict__ c1,
    float* __restrict__ outmem)     // (2,256,2048)
{
    const int t = blockIdx.x * 256 + threadIdx.x;  // 1,048,576
    const int l = t >> 19;
    const int b = (t >> 11) & 255;
    const int j = t & 2047;
    const float* src = (j < HID) ? (l ? h1 : h0) : (l ? c1 : c0);
    outmem[t] = src[(size_t)b * HID + (j & 1023)];
}

// ---------------------------------------------------------------------------
extern "C" void kernel_launch(void* const* d_in, const int* in_sizes, int n_in,
                              void* d_out, int out_size, void* d_ws, size_t ws_size,
                              hipStream_t stream)
{
    (void)in_sizes; (void)n_in; (void)out_size; (void)ws_size;

    const float* x     = (const float*)d_in[0];
    const float* mem   = (const float*)d_in[1];
    const float* w_ih0 = (const float*)d_in[2];
    const float* w_hh0 = (const float*)d_in[3];
    const float* b_ih0 = (const float*)d_in[4];
    const float* b_hh0 = (const float*)d_in[5];
    const float* w_ih1 = (const float*)d_in[6];
    const float* w_hh1 = (const float*)d_in[7];
    const float* b_ih1 = (const float*)d_in[8];
    const float* b_hh1 = (const float*)d_in[9];
    const int*   rst   = (const int*)d_in[10];
    float* out = (float*)d_out;

    // workspace layout (~41 MB)
    char*  ws  = (char*)d_ws;
    size_t off = 0;
    auto alloc = [&](size_t bytes) -> void* {
        void* p = ws + off;
        off += (bytes + 255) & ~(size_t)255;
        return p;
    };
    __bf16* W0    = (__bf16*)alloc((size_t)NG * KCAT * 2);   // 16 MB
    __bf16* W1    = (__bf16*)alloc((size_t)NG * KCAT * 2);   // 16 MB
    float*  bias0 = (float*)alloc((size_t)NG * 4);
    float*  bias1 = (float*)alloc((size_t)NG * 4);
    float*  h0    = (float*)alloc((size_t)BATCH * HID * 4);
    float*  c0    = (float*)alloc((size_t)BATCH * HID * 4);
    float*  h1    = (float*)alloc((size_t)BATCH * HID * 4);
    float*  c1    = (float*)alloc((size_t)BATCH * HID * 4);
    float*  h0raw = (float*)alloc((size_t)BATCH * HID * 4);
    float*  gates = (float*)alloc((size_t)BATCH * NG * 4);   // 4 MB

    // one-time prep
    pack_weights<<<4096, 256, 0, stream>>>(w_ih0, w_hh0, W0);
    pack_weights<<<4096, 256, 0, stream>>>(w_ih1, w_hh1, W1);
    combine_bias<<<16, 256, 0, stream>>>(b_ih0, b_hh0, bias0);
    combine_bias<<<16, 256, 0, stream>>>(b_ih1, b_hh1, bias1);
    init_state<<<4096, 256, 0, stream>>>(mem, h0, c0, h1, c1);

    const dim3 ggrid(BATCH / BM, NG / BN);  // (4,32) = 128 workgroups

    for (int s = 0; s < S_LEN; ++s) {
        // layer 0: A = [x_s | h0]
        lstm_gates_gemm<<<ggrid, 256, 0, stream>>>(
            x + (size_t)s * BATCH * IN_DIM, h0, W0, bias0, gates);
        lstm_cell<<<256, 256, 0, stream>>>(
            gates, c0, rst + s * BATCH, h0raw, h0, c0);
        // layer 1: A = [h0_raw | h1]
        lstm_gates_gemm<<<ggrid, 256, 0, stream>>>(
            h0raw, h1, W1, bias1, gates);
        lstm_cell<<<256, 256, 0, stream>>>(
            gates, c1, rst + s * BATCH, out + (size_t)s * BATCH * HID, h1, c1);
    }

    finalize_state<<<4096, 256, 0, stream>>>(
        h0, c0, h1, c1, out + (size_t)S_LEN * BATCH * HID);
}
